// NeighborModel_39058432589860
// MI455X (gfx1250) — compile-verified, matrix operand order checked by reading, not run on hardware
//
#include <hip/hip_runtime.h>
#include <hip/hip_bf16.h>

// ---------- types for WMMA ----------
typedef __attribute__((ext_vector_type(16))) __bf16       v16bf;
typedef __attribute__((ext_vector_type(8)))  float        v8f;
typedef __attribute__((ext_vector_type(4)))  unsigned int v4u;

#define DTOK 1222
#define KPAD 1248   // 39 * 32, zero-padded K for bf16 WMMA
#define NB   80
#define MTOK 160    // 2 * 80 tokens
#define FFD  2048
#define D3   3666   // 3 * DTOK
#define HH   224

// ---------- helpers ----------
__device__ __forceinline__ float block_reduce_sum(float v, float* sm) {
    int tid = threadIdx.x;
    sm[tid] = v;
    __syncthreads();
    for (int s = 128; s > 0; s >>= 1) {
        if (tid < s) sm[tid] += sm[tid + s];
        __syncthreads();
    }
    float r = sm[0];
    __syncthreads();
    return r;
}

// ---------- 2x2 average pool (pyramid build) ----------
__global__ __launch_bounds__(256) void avgpool2_kernel(
    const float* __restrict__ src, float* __restrict__ dst, int BC, int Hd, int Wd) {
    int total = BC * Hd * Wd;
    for (int i = blockIdx.x * blockDim.x + threadIdx.x; i < total; i += gridDim.x * blockDim.x) {
        int w  = i % Wd;
        int h  = (i / Wd) % Hd;
        int bc = i / (Wd * Hd);
        const float* s = src + ((size_t)bc * (2 * Hd) + 2 * h) * (size_t)(2 * Wd) + 2 * w;
        dst[i] = 0.25f * (s[0] + s[1] + s[2 * Wd] + s[2 * Wd + 1]);
    }
}

// ---------- fp32 -> bf16 with K zero-padding ----------
__global__ __launch_bounds__(256) void cvt_pad_bf16_kernel(
    const float* __restrict__ w, __bf16* __restrict__ o, int rows, int K, int Kp) {
    int total = rows * Kp;
    for (int i = blockIdx.x * blockDim.x + threadIdx.x; i < total; i += gridDim.x * blockDim.x) {
        int k = i % Kp;
        int r = i / Kp;
        o[i] = (k < K) ? (__bf16)w[(size_t)r * K + k] : (__bf16)0.0f;
    }
}

__global__ void copy_int_kernel(const int* __restrict__ a, int* __restrict__ b, int n) {
    int i = blockIdx.x * blockDim.x + threadIdx.x;
    if (i < n) b[i] = a[i];
}

// ---------- token build: gather + pyramid dots + LN + PE ----------
__global__ __launch_bounds__(256) void build_tokens_kernel(
    const float* __restrict__ feat,     // (2,1024,224,224)
    const float* __restrict__ pyr1,     // (2,1024,112,112)
    const float* __restrict__ pyr2,     // (2,1024,56,56)
    const float* __restrict__ pyr3,     // (2,1024,28,28)
    const int*   __restrict__ boundary, // (2,80,2)
    const float* __restrict__ ln_g, const float* __restrict__ ln_b,
    float*  __restrict__ tok_f32,       // (160,1222)
    __bf16* __restrict__ tok_bf) {      // (160,1248) zero-padded
    __shared__ float qf[1024];
    __shared__ float tok[DTOK];
    __shared__ float sm[256];
    int t = blockIdx.x;                 // token id 0..159
    int b = t / NB, n = t % NB;
    int bx = boundary[t * 2 + 0], by = boundary[t * 2 + 1];
    int tid = threadIdx.x;
    for (int c = tid; c < 1024; c += 256) {
        float v = feat[(((size_t)b * 1024 + c) * HH + bx) * HH + by];
        qf[c] = v;
        tok[c] = v;
    }
    __syncthreads();
    // 196 neighborhood dot products (4 scales x 49 offsets), C=1024
    for (int d = tid; d < 196; d += 256) {
        int s  = d / 49, o = d % 49;
        int dx = o / 7 - 3, dy = o % 7 - 3;
        int Hs = HH >> s;
        int r  = (bx >> s) + dx;
        int cc = (by >> s) + dy;
        float acc = 0.f;
        if (r >= 0 && r < Hs && cc >= 0 && cc < Hs) {
            const float* p = (s == 0) ? feat : (s == 1) ? pyr1 : (s == 2) ? pyr2 : pyr3;
            const float* pp = p + (((size_t)b * 1024) * Hs + r) * (size_t)Hs + cc;
            size_t cstr = (size_t)Hs * Hs;
            for (int c = 0; c < 1024; ++c) acc += qf[c] * pp[(size_t)c * cstr];
        }
        tok[1024 + d] = acc;
    }
    if (tid == 0) { tok[1220] = (float)bx; tok[1221] = (float)by; }
    __syncthreads();
    // LayerNorm over D, then add positional encoding
    float ps = 0.f, ps2 = 0.f;
    for (int i = tid; i < DTOK; i += 256) { float v = tok[i]; ps += v; ps2 += v * v; }
    float sum  = block_reduce_sum(ps, sm);
    float sum2 = block_reduce_sum(ps2, sm);
    float mean = sum / (float)DTOK;
    float var  = sum2 / (float)DTOK - mean * mean;
    float rinv = rsqrtf(var + 1e-5f);
    for (int i = tid; i < KPAD; i += 256) {
        if (i < DTOK) {
            int   j2 = (i >> 1) << 1;
            float f  = __expf(-logf(10000.f) * (float)j2 / (float)DTOK);
            float pe = (i & 1) ? __cosf((float)n * f) : __sinf((float)n * f);
            float v  = (tok[i] - mean) * rinv * ln_g[i] + ln_b[i] + pe;
            tok_f32[(size_t)t * DTOK + i] = v;
            tok_bf [(size_t)t * KPAD + i] = (__bf16)v;
        } else {
            tok_bf[(size_t)t * KPAD + i] = (__bf16)0.0f;
        }
    }
}

// ---------- generic bf16 WMMA GEMM: C[M,N] = A[M,K] * W[N,K]^T + bias (+res, relu) ----------
__global__ __launch_bounds__(256) void gemm_bf16_wmma(
    const __bf16* __restrict__ A, int lda,        // M x lda (K padded)
    const __bf16* __restrict__ W, int ldw,        // N x ldw row-major weights
    const float*  __restrict__ bias,
    const float*  __restrict__ residual, int ldr, // nullable
    float*  __restrict__ Cf, int ldc,             // nullable f32 output
    __bf16* __restrict__ Cb, int ldcb,            // nullable bf16 output
    int M, int N, int K, int relu) {
    int lane = threadIdx.x & 31;
    int wid  = threadIdx.x >> 5;
    int tile_n = blockIdx.x * 8 + wid;
    int tile_m = blockIdx.y;
    int nTiles = (N + 15) >> 4;
    if (tile_n >= nTiles) return;

    int halfsel = lane >> 4;      // K-group select (ISA 16-bit A/B wave32 layout)
    int lidx    = lane & 15;
    int row = tile_m * 16 + lidx; // A row for this lane's fragment
    int col = tile_n * 16 + lidx; // output column == weight row
    int colc = (col < N) ? col : (N - 1);

    const char* arow = (const char*)(A + (size_t)row  * lda);
    const char* wrow = (const char*)(W + (size_t)colc * ldw);

    v8f acc = {};
    for (int k0 = 0; k0 < K; k0 += 32) {
        int kb = k0 + halfsel * 8;
        union { v16bf v; v4u u[2]; } a, bm;
        a.u[0]  = *(const v4u*)(arow + 2 * (size_t)kb);
        a.u[1]  = *(const v4u*)(arow + 2 * (size_t)(kb + 16));
        bm.u[0] = *(const v4u*)(wrow + 2 * (size_t)kb);
        bm.u[1] = *(const v4u*)(wrow + 2 * (size_t)(kb + 16));
        acc = __builtin_amdgcn_wmma_f32_16x16x32_bf16(
            false, a.v, false, bm.v, (short)0, acc, false, false);
    }
    if (col < N) {
        float bv = bias ? bias[col] : 0.f;
#pragma unroll
        for (int r = 0; r < 8; ++r) {
            int orow = tile_m * 16 + r + 8 * halfsel;
            if (orow < M) {
                float v = acc[r] + bv;
                if (residual) v += residual[(size_t)orow * ldr + col];
                if (relu) v = v > 0.f ? v : 0.f;
                if (Cf) Cf[(size_t)orow * ldc  + col] = v;
                if (Cb) Cb[(size_t)orow * ldcb + col] = (__bf16)v;
            }
        }
    }
}

// ---------- single-head attention over 80 tokens ----------
__global__ __launch_bounds__(256) void attention_kernel(
    const float* __restrict__ qkv,    // (160, 3666) : [q | k | v]
    __bf16* __restrict__ attn_bf) {   // (160, 1248) zero-padded
    __shared__ float sc[NB];
    int t = blockIdx.x;
    int b = t / NB;
    int tid = threadIdx.x, lane = tid & 31, wid = tid >> 5;
    const float* q = qkv + (size_t)t * D3;
    float scal = rsqrtf((float)DTOK);
    for (int m = wid; m < NB; m += 8) {
        const float* k = qkv + (size_t)(b * NB + m) * D3 + DTOK;
        float acc = 0.f;
        for (int d = lane; d < DTOK; d += 32) acc += q[d] * k[d];
        for (int off = 16; off > 0; off >>= 1) acc += __shfl_xor(acc, off, 32);
        if (lane == 0) sc[m] = acc * scal;
    }
    __syncthreads();
    if (tid == 0) {
        float mx = sc[0];
        for (int m = 1; m < NB; ++m) mx = fmaxf(mx, sc[m]);
        float s = 0.f;
        for (int m = 0; m < NB; ++m) { sc[m] = __expf(sc[m] - mx); s += sc[m]; }
        float inv = 1.f / s;
        for (int m = 0; m < NB; ++m) sc[m] *= inv;
    }
    __syncthreads();
    for (int d = tid; d < KPAD; d += 256) {
        if (d < DTOK) {
            float acc = 0.f;
            const float* v = qkv + (size_t)(b * NB) * D3 + 2 * DTOK + d;
            for (int m = 0; m < NB; ++m) acc += sc[m] * v[(size_t)m * D3];
            attn_bf[(size_t)t * KPAD + d] = (__bf16)acc;
        } else {
            attn_bf[(size_t)t * KPAD + d] = (__bf16)0.0f;
        }
    }
}

// ---------- LayerNorm (f32 in, f32 and/or padded bf16 out) ----------
__global__ __launch_bounds__(256) void ln_kernel(
    const float* __restrict__ x, const float* __restrict__ g, const float* __restrict__ bta,
    float* __restrict__ outf, __bf16* __restrict__ outb) {
    __shared__ float sm[256];
    int t = blockIdx.x, tid = threadIdx.x;
    const float* row = x + (size_t)t * DTOK;
    float ps = 0.f, ps2 = 0.f;
    for (int i = tid; i < DTOK; i += 256) { float v = row[i]; ps += v; ps2 += v * v; }
    float sum  = block_reduce_sum(ps, sm);
    float sum2 = block_reduce_sum(ps2, sm);
    float mean = sum / (float)DTOK;
    float var  = sum2 / (float)DTOK - mean * mean;
    float rinv = rsqrtf(var + 1e-5f);
    for (int i = tid; i < KPAD; i += 256) {
        if (i < DTOK) {
            float v = (row[i] - mean) * rinv * g[i] + bta[i];
            if (outf) outf[(size_t)t * DTOK + i] = v;
            if (outb) outb[(size_t)t * KPAD + i] = (__bf16)v;
        } else if (outb) {
            outb[(size_t)t * KPAD + i] = (__bf16)0.0f;
        }
    }
}

// ---------- fc head: only out[..., :2] is ever used ----------
__global__ __launch_bounds__(256) void fc_update_kernel(
    const float* __restrict__ x,    // (160,1222)
    const float* __restrict__ fcw,  // (80,1026,1222)
    const float* __restrict__ fcb,  // (80,1026)
    int* __restrict__ boundary,     // (2,80,2) in/out
    int* __restrict__ out_it) {     // (2,80,2) this iteration's result
    __shared__ float sm[256];
    int t = blockIdx.x, tid = threadIdx.x;
    int n = t % NB;
    const float* row = x + (size_t)t * DTOK;
    const float* w0  = fcw + ((size_t)n * 1026 + 0) * DTOK;
    const float* w1  = fcw + ((size_t)n * 1026 + 1) * DTOK;
    float a0 = 0.f, a1 = 0.f;
    for (int i = tid; i < DTOK; i += 256) { float v = row[i]; a0 += v * w0[i]; a1 += v * w1[i]; }
    float s0 = block_reduce_sum(a0, sm);
    float s1 = block_reduce_sum(a1, sm);
    if (tid == 0) {
        float o0 = s0 + fcb[(size_t)n * 1026 + 0];
        float o1 = s1 + fcb[(size_t)n * 1026 + 1];
        int b0 = boundary[t * 2 + 0] + (int)truncf(o0);
        int b1 = boundary[t * 2 + 1] + (int)truncf(o1);
        b0 = min(max(b0, 0), 223);
        b1 = min(max(b1, 0), 223);
        boundary[t * 2 + 0] = b0;
        boundary[t * 2 + 1] = b1;
        out_it[t * 2 + 0] = b0;
        out_it[t * 2 + 1] = b1;
    }
}

extern "C" void kernel_launch(void* const* d_in, const int* in_sizes, int n_in,
                              void* d_out, int out_size, void* d_ws, size_t ws_size,
                              hipStream_t stream) {
    const float* feat      = (const float*)d_in[0];
    const int*   bnd_in    = (const int*)  d_in[1];
    const float* ln_g      = (const float*)d_in[2];
    const float* ln_b      = (const float*)d_in[3];
    const float* inproj_w  = (const float*)d_in[4];
    const float* inproj_b  = (const float*)d_in[5];
    const float* outproj_w = (const float*)d_in[6];
    const float* outproj_b = (const float*)d_in[7];
    const float* lin1_w    = (const float*)d_in[8];
    const float* lin1_b    = (const float*)d_in[9];
    const float* lin2_w    = (const float*)d_in[10];
    const float* lin2_b    = (const float*)d_in[11];
    const float* n1g       = (const float*)d_in[12];
    const float* n1b       = (const float*)d_in[13];
    const float* n2g       = (const float*)d_in[14];
    const float* n2b       = (const float*)d_in[15];
    const float* fcw       = (const float*)d_in[16];
    const float* fcb       = (const float*)d_in[17];
    int* outp = (int*)d_out; // (6,2,80,2) int32 boundaries

    // workspace bump allocator (256B aligned)
    char*  ws  = (char*)d_ws;
    size_t off = 0;
    auto alloc = [&](size_t bytes) -> char* {
        char* p = ws + off;
        off += (bytes + 255) & ~(size_t)255;
        return p;
    };
    float*  pyr1   = (float*) alloc((size_t)2 * 1024 * 112 * 112 * 4);
    float*  pyr2   = (float*) alloc((size_t)2 * 1024 * 56  * 56  * 4);
    float*  pyr3   = (float*) alloc((size_t)2 * 1024 * 28  * 28  * 4);
    __bf16* w_inp  = (__bf16*)alloc((size_t)D3   * KPAD * 2);
    __bf16* w_outp = (__bf16*)alloc((size_t)DTOK * KPAD * 2);
    __bf16* w_l1   = (__bf16*)alloc((size_t)FFD  * KPAD * 2);
    __bf16* w_l2   = (__bf16*)alloc((size_t)DTOK * FFD  * 2);
    float*  tok_f  = (float*) alloc((size_t)MTOK * DTOK * 4);
    __bf16* tok_bf = (__bf16*)alloc((size_t)MTOK * KPAD * 2);
    float*  qkv    = (float*) alloc((size_t)MTOK * D3   * 4);
    __bf16* att_bf = (__bf16*)alloc((size_t)MTOK * KPAD * 2);
    float*  xpre   = (float*) alloc((size_t)MTOK * DTOK * 4);
    float*  x_f    = (float*) alloc((size_t)MTOK * DTOK * 4);
    __bf16* x_bf   = (__bf16*)alloc((size_t)MTOK * KPAD * 2);
    __bf16* ff1_bf = (__bf16*)alloc((size_t)MTOK * FFD  * 2);
    float*  ypre   = (float*) alloc((size_t)MTOK * DTOK * 4);
    float*  xf     = (float*) alloc((size_t)MTOK * DTOK * 4);
    int*    bnd    = (int*)   alloc((size_t)MTOK * 2 * 4);

    dim3 blk(256);
    // boundary init
    copy_int_kernel<<<1, 320, 0, stream>>>(bnd_in, bnd, 320);
    // pyramid build (once per launch)
    {
        int t1 = 2048 * 112 * 112;
        avgpool2_kernel<<<(t1 + 255) / 256, blk, 0, stream>>>(feat, pyr1, 2048, 112, 112);
        int t2 = 2048 * 56 * 56;
        avgpool2_kernel<<<(t2 + 255) / 256, blk, 0, stream>>>(pyr1, pyr2, 2048, 56, 56);
        int t3 = 2048 * 28 * 28;
        avgpool2_kernel<<<(t3 + 255) / 256, blk, 0, stream>>>(pyr2, pyr3, 2048, 28, 28);
    }
    // weight conversion to zero-padded bf16
    {
        int t;
        t = D3 * KPAD;   cvt_pad_bf16_kernel<<<(t + 255) / 256, blk, 0, stream>>>(inproj_w,  w_inp,  D3,   DTOK, KPAD);
        t = DTOK * KPAD; cvt_pad_bf16_kernel<<<(t + 255) / 256, blk, 0, stream>>>(outproj_w, w_outp, DTOK, DTOK, KPAD);
        t = FFD * KPAD;  cvt_pad_bf16_kernel<<<(t + 255) / 256, blk, 0, stream>>>(lin1_w,    w_l1,   FFD,  DTOK, KPAD);
        t = DTOK * FFD;  cvt_pad_bf16_kernel<<<(t + 255) / 256, blk, 0, stream>>>(lin2_w,    w_l2,   DTOK, FFD,  FFD);
    }

    for (int it = 0; it < 6; ++it) {
        build_tokens_kernel<<<MTOK, blk, 0, stream>>>(
            feat, pyr1, pyr2, pyr3, bnd, ln_g, ln_b, tok_f, tok_bf);
        // qkv = tokens @ in_proj_w^T + b
        {
            int N = D3, nT = (N + 15) / 16;
            dim3 g((nT + 7) / 8, MTOK / 16);
            gemm_bf16_wmma<<<g, blk, 0, stream>>>(tok_bf, KPAD, w_inp, KPAD, inproj_b,
                                                  nullptr, 0, qkv, D3, nullptr, 0,
                                                  MTOK, N, KPAD, 0);
        }
        attention_kernel<<<MTOK, blk, 0, stream>>>(qkv, att_bf);
        // xpre = tokens + attn @ out_proj_w^T + b
        {
            int N = DTOK, nT = (N + 15) / 16;
            dim3 g((nT + 7) / 8, MTOK / 16);
            gemm_bf16_wmma<<<g, blk, 0, stream>>>(att_bf, KPAD, w_outp, KPAD, outproj_b,
                                                  tok_f, DTOK, xpre, DTOK, nullptr, 0,
                                                  MTOK, N, KPAD, 0);
        }
        ln_kernel<<<MTOK, blk, 0, stream>>>(xpre, n1g, n1b, x_f, x_bf);
        // ff1 = relu(x @ lin1_w^T + b) -> bf16
        {
            int N = FFD, nT = (N + 15) / 16;
            dim3 g((nT + 7) / 8, MTOK / 16);
            gemm_bf16_wmma<<<g, blk, 0, stream>>>(x_bf, KPAD, w_l1, KPAD, lin1_b,
                                                  nullptr, 0, nullptr, 0, ff1_bf, FFD,
                                                  MTOK, N, KPAD, 1);
        }
        // ypre = x + ff1 @ lin2_w^T + b
        {
            int N = DTOK, nT = (N + 15) / 16;
            dim3 g((nT + 7) / 8, MTOK / 16);
            gemm_bf16_wmma<<<g, blk, 0, stream>>>(ff1_bf, FFD, w_l2, FFD, lin2_b,
                                                  x_f, DTOK, ypre, DTOK, nullptr, 0,
                                                  MTOK, N, FFD, 0);
        }
        ln_kernel<<<MTOK, blk, 0, stream>>>(ypre, n2g, n2b, xf, nullptr);
        fc_update_kernel<<<MTOK, blk, 0, stream>>>(xf, fcw, fcb, bnd, outp + it * 320);
    }
}
